// LNN_54949811585278
// MI455X (gfx1250) — matrix operand order repaired and checked by reading, not compiled, not run on hardware
//
#include <hip/hip_runtime.h>
#include <hip/hip_bf16.h>

typedef __attribute__((ext_vector_type(16))) _Float16 v16h;
typedef __attribute__((ext_vector_type(8)))  _Float16 v8h;
typedef __attribute__((ext_vector_type(8)))  float    v8f;

// ---------------------------------------------------------------------------
// Pack conv weights (OIHW) into the exact WMMA B-operand per-lane register
// image, f16, K zero-padded to a multiple of 32.
// K-order is PERMUTED to (ky, kx, ci):  k = (ky*KS + kx)*CIN + ci
// so that A-side groups of 8 consecutive k walk the channel dim of one pixel
// (vectorizable against NHWC activations).
// Layout: wpack[((chunk*NT + ntile)*32 + lane)*16 + e]
//   n = ntile*16 + (lane&15),  k = chunk*32 + (lane>>4)*8 + (e<8 ? e : e+8)
// ---------------------------------------------------------------------------
__global__ void pack_b_kernel(const float* __restrict__ w, _Float16* __restrict__ wpack,
                              int CIN, int KS, int COUT, int K, int TOTAL)
{
    int idx = blockIdx.x * blockDim.x + threadIdx.x;
    if (idx >= TOTAL) return;
    int e  = idx & 15;
    int l  = (idx >> 4) & 31;
    int t  = idx >> 9;
    int NT = COUT >> 4;
    int nt = t % NT;
    int c  = t / NT;
    int n  = nt * 16 + (l & 15);
    int k  = c * 32 + (l >> 4) * 8 + (e < 8 ? e : e + 8);
    float v = 0.0f;
    if (k < K) {
        int cell = k / CIN;
        int ci   = k - cell * CIN;
        int ky   = cell / KS;
        int kx   = cell - ky * KS;
        v = w[((n * CIN + ci) * KS + ky) * KS + kx];
    }
    wpack[idx] = (_Float16)v;
}

// ---------------------------------------------------------------------------
// Implicit-GEMM conv via V_WMMA_F32_16X16X32_F16, NHWC activations (f16).
// One wave per 16-row M-tile (M = B*OH*OW), covering all N-tiles (A reuse).
// A: for CIN%8==0, each 8-element register group = one 16B vector load of a
//    single input pixel's channels (one bounds check); else scalar gather.
// B: one aligned 32B vector load per N-tile from the pre-packed lane image.
// Fused bias + optional BN + ReLU epilogue; coalesced NHWC f16 stores.
// Grids sized exactly -> EXEC all-ones at every WMMA.
// ---------------------------------------------------------------------------
template<int CIN, int COUT, int KS, int STRIDE, int PAD,
         int HIN, int WIN, int HOUT, int WOUT, bool INF32, bool DOBN>
__global__ __launch_bounds__(128) void conv_wmma_kernel(
    const void* __restrict__ inpv, const _Float16* __restrict__ wpack,
    const float* __restrict__ bias,
    const float* __restrict__ bn_g, const float* __restrict__ bn_b,
    const float* __restrict__ bn_m, const float* __restrict__ bn_v,
    _Float16* __restrict__ outp)
{
    constexpr int  K   = CIN * KS * KS;
    constexpr int  KCH = (K + 31) / 32;
    constexpr int  NT  = COUT / 16;
    constexpr int  HW  = HOUT * WOUT;
    constexpr bool VEC = (CIN % 8) == 0;

    const int lane  = threadIdx.x & 31;
    const int mtile = blockIdx.x * (blockDim.x >> 5) + (threadIdx.x >> 5);

    // A-row coordinates for this lane (16-bit A layout: lanes 0-15 / 16-31)
    const int m   = mtile * 16 + (lane & 15);
    const int nb  = m / HW;                 // HW is 2^n for conv2-4 -> shifts
    const int rem = m - nb * HW;
    const int oy  = rem / WOUT;
    const int ox  = rem - oy * WOUT;
    const int iy0 = oy * STRIDE - PAD;
    const int ix0 = ox * STRIDE - PAD;
    const int khi = lane >> 4;

    const float*    inF = (const float*)inpv;
    const _Float16* inH = (const _Float16*)inpv;

    v8f acc[NT] = {};

    for (int c = 0; c < KCH; ++c) {
        // ---- gather A tile (16x32 f16) into WMMA lane layout ----
        v16h a = {};
        if (VEC) {
#pragma unroll
            for (int g = 0; g < 2; ++g) {
                int kk = c * 32 + khi * 8 + g * 16;   // 8 consecutive k = one pixel's channels
                if (kk < K) {
                    int cell = kk / CIN;              // CIN = 2^n -> shift
                    int cio  = kk - cell * CIN;
                    int ky   = cell / KS;
                    int kx   = cell - ky * KS;
                    int iy = iy0 + ky, ix = ix0 + kx;
                    if (iy >= 0 && iy < HIN && ix >= 0 && ix < WIN) {
                        v8h vv = *(const v8h*)(inH + ((nb * HIN + iy) * WIN + ix) * CIN + cio);
#pragma unroll
                        for (int i = 0; i < 8; ++i) a[g * 8 + i] = vv[i];
                    }
                }
            }
        } else {
#pragma unroll
            for (int e = 0; e < 16; ++e) {
                int k = c * 32 + khi * 8 + (e < 8 ? e : e + 8);
                if (k < K) {
                    int cell = k / CIN;
                    int cio  = k - cell * CIN;
                    int ky   = cell / KS;
                    int kx   = cell - ky * KS;
                    int iy = iy0 + ky, ix = ix0 + kx;
                    if (iy >= 0 && iy < HIN && ix >= 0 && ix < WIN) {
                        int idx = ((nb * HIN + iy) * WIN + ix) * CIN + cio;
                        a[e] = INF32 ? (_Float16)inF[idx] : inH[idx];
                    }
                }
            }
        }
        // ---- B tiles: one 32B vector load each (pre-packed lane image) ----
        const _Float16* bp = wpack + ((long)c * NT) * 512 + lane * 16;
#pragma unroll
        for (int nt = 0; nt < NT; ++nt) {
            v16h bmat = *(const v16h*)(bp + nt * 512);
            acc[nt] = __builtin_amdgcn_wmma_f32_16x16x32_f16(
                false, a, false, bmat, (short)0, acc[nt], false, false);
        }
        if (c + 1 < KCH) __builtin_prefetch(bp + NT * 512, 0, 1);
    }

    // ---- epilogue: bias (+BN) + ReLU, coalesced NHWC f16 stores ----
    const int mmbase = mtile * 16 + khi * 8;
#pragma unroll
    for (int nt = 0; nt < NT; ++nt) {
        int nn = nt * 16 + (lane & 15);   // output channel (consecutive on lanes)
        float sc = 1.0f, sh = 0.0f;
        if (DOBN) {
            float inv = rsqrtf(bn_v[nn] + 1e-5f);
            sc = bn_g[nn] * inv;
            sh = bn_b[nn] - bn_m[nn] * sc;
        }
        float bsv = bias[nn];
#pragma unroll
        for (int r = 0; r < 8; ++r) {
            int mm  = mmbase + r;
            float v = acc[nt][r] + bsv;
            if (DOBN) v = v * sc + sh;
            v = fmaxf(v, 0.0f);
            int nb2 = mm / HW;
            int rm2 = mm - nb2 * HW;      // = oy*WOUT + ox
            outp[(nb2 * HW + rm2) * COUT + nn] = (_Float16)v;
        }
    }
}

// ---------------------------------------------------------------------------
// 2x2 maxpool on NHWC (512,8,8,128) f16 -> NCHW-pooled f32 flat
// [512][128][4][4]; flat order == reference reshape to [512][64][32].
// ---------------------------------------------------------------------------
__global__ void pool_kernel(const _Float16* __restrict__ a4, float* __restrict__ seq)
{
    int idx = blockIdx.x * blockDim.x + threadIdx.x;
    if (idx >= 512 * 128 * 16) return;
    int pp  = idx & 15;
    int t   = idx >> 4;
    int cch = t & 127;
    int b   = t >> 7;
    int py = pp >> 2, px = pp & 3;
    int base = ((b * 8 + py * 2) * 8 + px * 2) * 128 + cch;   // NHWC
    float v0 = (float)a4[base],        v1 = (float)a4[base + 128];
    float v2 = (float)a4[base + 1024], v3 = (float)a4[base + 1152];
    seq[b * 2048 + cch * 16 + pp] = fmaxf(fmaxf(v0, v1), fmaxf(v2, v3));
}

// ---------------------------------------------------------------------------
// CfC recurrence: one wave32 per batch sample, 8 waves / block.
// Pre-masked weights resident in LDS (~57 KB) shared by the block.
// Lane j computes output neuron j's four dot products per step; z broadcast
// from LDS (conflict-free), weight rows at strides 65/54/31 dwords
// (conflict-free mod 64 banks). h-state layout h0|h1|h2 makes layer-1/2
// z-vectors contiguous slices of hbuf.
// ---------------------------------------------------------------------------
__device__ __forceinline__ float sigm(float x) { return 1.0f / (1.0f + __expf(-x)); }

template<int H, int TOT>
__device__ __forceinline__ void cfc_compute(const float* __restrict__ wl,
                                            const float* __restrict__ bl,
                                            const float* z, float* hdst, int lane)
{
    constexpr int NP = (H + 31) / 32;
#pragma unroll
    for (int pass = 0; pass < NP; ++pass) {
        int j  = lane + pass * 32;
        int jj = (j < H) ? j : (H - 1);
        float a1 = bl[jj], a2 = bl[H + jj], a3 = bl[2 * H + jj], a4 = bl[3 * H + jj];
        const float* w1 = wl + jj * TOT;
        const float* w2 = wl + H * TOT + jj * TOT;
        const float* w3 = wl + 2 * H * TOT + jj * TOT;
        const float* w4 = wl + 3 * H * TOT + jj * TOT;
        for (int k = 0; k < TOT; ++k) {
            float zk = z[k];
            a1 = fmaf(w1[k], zk, a1);
            a2 = fmaf(w2[k], zk, a2);
            a3 = fmaf(w3[k], zk, a3);
            a4 = fmaf(w4[k], zk, a4);
        }
        float f1 = tanhf(a1), f2 = tanhf(a2);
        float ti = sigm(a3 + a4);
        float hn = f1 * (1.0f - ti) + ti * f2;
        if (j < H) hdst[j] = hn;   // stores issue after all wave-wide z reads
    }
}

__global__ __launch_bounds__(256) void cfc_rnn_kernel(
    const float* __restrict__ seq,
    const float* W0a, const float* B0a, const float* W0b, const float* B0b,
    const float* W0c, const float* B0c, const float* W0d, const float* B0d,
    const float* W1a, const float* B1a, const float* W1b, const float* B1b,
    const float* W1c, const float* B1c, const float* W1d, const float* B1d,
    const float* W2a, const float* B2a, const float* W2b, const float* B2b,
    const float* W2c, const float* B2c, const float* W2d, const float* B2d,
    const float* M0, const float* M1, const float* M2,
    float* __restrict__ out)
{
    __shared__ float w_all[14356];   // L0:4x2145 @0, L1:4x1134 @8580, L2:4x310 @13116
    __shared__ float b_all[256];     // L0 @0 (4x33), L1 @132 (4x21), L2 @216 (4x10)
    __shared__ float hbuf[8][80];    // per-wave: h0[0..32] h1[33..53] h2[54..63]
    __shared__ float zbuf[8][72];    // per-wave layer-0 z (len 65)

    const int tid = threadIdx.x;
    for (int i = tid; i < 2145; i += 256) {
        float mk = M0[i];
        w_all[i]          = W0a[i] * mk;
        w_all[2145 + i]   = W0b[i] * mk;
        w_all[4290 + i]   = W0c[i];
        w_all[6435 + i]   = W0d[i];
    }
    for (int i = tid; i < 1134; i += 256) {
        float mk = M1[i];
        w_all[8580 + i]   = W1a[i] * mk;
        w_all[9714 + i]   = W1b[i] * mk;
        w_all[10848 + i]  = W1c[i];
        w_all[11982 + i]  = W1d[i];
    }
    for (int i = tid; i < 310; i += 256) {
        float mk = M2[i];
        w_all[13116 + i]  = W2a[i] * mk;
        w_all[13426 + i]  = W2b[i] * mk;
        w_all[13736 + i]  = W2c[i];
        w_all[14046 + i]  = W2d[i];
    }
    if (tid < 33) { b_all[tid] = B0a[tid]; b_all[33 + tid] = B0b[tid];
                    b_all[66 + tid] = B0c[tid]; b_all[99 + tid] = B0d[tid]; }
    if (tid < 21) { b_all[132 + tid] = B1a[tid]; b_all[153 + tid] = B1b[tid];
                    b_all[174 + tid] = B1c[tid]; b_all[195 + tid] = B1d[tid]; }
    if (tid < 10) { b_all[216 + tid] = B2a[tid]; b_all[226 + tid] = B2b[tid];
                    b_all[236 + tid] = B2c[tid]; b_all[246 + tid] = B2d[tid]; }

    const int w = tid >> 5, lane = tid & 31;
    hbuf[w][lane] = 0.0f;
    hbuf[w][lane + 32] = 0.0f;
    __syncthreads();

    const int b = blockIdx.x * 8 + w;
    const float* sb = seq + b * (64 * 32);

    for (int t = 0; t < 64; ++t) {
        // layer-0 z = [x(32) | h0(33)]
        zbuf[w][lane]      = sb[t * 32 + lane];
        zbuf[w][32 + lane] = hbuf[w][lane];
        if (lane == 0) zbuf[w][64] = hbuf[w][32];
        __syncthreads();
        cfc_compute<33, 65>(w_all + 0,     b_all + 0,   &zbuf[w][0],  &hbuf[w][0],  lane);
        __syncthreads();
        // layer-1 z = [h0(33) | h1(21)] == hbuf[0..53]
        cfc_compute<21, 54>(w_all + 8580,  b_all + 132, &hbuf[w][0],  &hbuf[w][33], lane);
        __syncthreads();
        // layer-2 z = [h1(21) | h2(10)] == hbuf[33..63]
        cfc_compute<10, 31>(w_all + 13116, b_all + 216, &hbuf[w][33], &hbuf[w][54], lane);
        __syncthreads();
    }
    if (lane < 10) out[b * 10 + lane] = hbuf[w][54 + lane];
}

// ---------------------------------------------------------------------------
extern "C" void kernel_launch(void* const* d_in, const int* in_sizes, int n_in,
                              void* d_out, int out_size, void* d_ws, size_t ws_size,
                              hipStream_t stream)
{
    const float* x   = (const float*)d_in[0];
    const float* w1  = (const float*)d_in[1];
    const float* b1  = (const float*)d_in[2];
    const float* w2  = (const float*)d_in[3];
    const float* b2c = (const float*)d_in[4];
    const float* w3  = (const float*)d_in[5];
    const float* b3  = (const float*)d_in[6];
    const float* w4  = (const float*)d_in[7];
    const float* b4  = (const float*)d_in[8];
    const float* g2  = (const float*)d_in[9];
    const float* bb2 = (const float*)d_in[10];
    const float* g4  = (const float*)d_in[11];
    const float* bb4 = (const float*)d_in[12];
    const float* m2  = (const float*)d_in[13];
    const float* v2  = (const float*)d_in[14];
    const float* m4  = (const float*)d_in[15];
    const float* v4  = (const float*)d_in[16];
    // rnn_params: per layer W_ff1,b_ff1,W_ff2,b_ff2,W_ta,b_ta,W_tb,b_tb
    const float* R[24];
    for (int i = 0; i < 24; ++i) R[i] = (const float*)d_in[17 + i];
    const float* M0 = (const float*)d_in[41];
    const float* M1 = (const float*)d_in[42];
    const float* M2 = (const float*)d_in[43];

    char* ws = (char*)d_ws;
    size_t off = 0;
    auto alloc = [&](size_t bytes) -> void* {
        void* p = (void*)(ws + off);
        off += bytes;
        off = (off + 255) & ~(size_t)255;
        return p;
    };
    _Float16* wp1  = (_Float16*)alloc((size_t)512 * 2);
    _Float16* wp2  = (_Float16*)alloc((size_t)5120 * 2);
    _Float16* wp3  = (_Float16*)alloc((size_t)51200 * 2);
    _Float16* wp4  = (_Float16*)alloc((size_t)204800 * 2);
    _Float16* act1 = (_Float16*)alloc((size_t)512 * 62 * 62 * 16 * 2);   // NHWC
    _Float16* act2 = (_Float16*)alloc((size_t)512 * 32 * 32 * 32 * 2);   // NHWC
    _Float16* act3 = (_Float16*)alloc((size_t)512 * 16 * 16 * 64 * 2);   // NHWC
    _Float16* act4 = (_Float16*)alloc((size_t)512 * 8 * 8 * 128 * 2);    // NHWC
    float*    seq  = (float*)   alloc((size_t)512 * 64 * 32 * 4);

    // ---- pack weights into WMMA B lane-layout (f16, permuted-K, padded) ----
    pack_b_kernel<<<(512 + 255) / 256, 256, 0, stream>>>(w1, wp1, 1, 3, 16, 9, 512);
    pack_b_kernel<<<(5120 + 255) / 256, 256, 0, stream>>>(w2, wp2, 16, 3, 32, 144, 5120);
    pack_b_kernel<<<(51200 + 255) / 256, 256, 0, stream>>>(w3, wp3, 32, 5, 64, 800, 51200);
    pack_b_kernel<<<(204800 + 255) / 256, 256, 0, stream>>>(w4, wp4, 64, 5, 128, 1600, 204800);

    // ---- conv chain (implicit GEMM, WMMA f16->f32, NHWC) ----
    conv_wmma_kernel<1, 16, 3, 1, 0, 64, 64, 62, 62, true, false>
        <<<30752, 128, 0, stream>>>(x, wp1, b1, nullptr, nullptr, nullptr, nullptr, act1);
    conv_wmma_kernel<16, 32, 3, 2, 2, 62, 62, 32, 32, false, true>
        <<<8192, 128, 0, stream>>>(act1, wp2, b2c, g2, bb2, m2, v2, act2);
    conv_wmma_kernel<32, 64, 5, 2, 2, 32, 32, 16, 16, false, false>
        <<<2048, 128, 0, stream>>>(act2, wp3, b3, nullptr, nullptr, nullptr, nullptr, act3);
    conv_wmma_kernel<64, 128, 5, 2, 2, 16, 16, 8, 8, false, true>
        <<<512, 128, 0, stream>>>(act3, wp4, b4, g4, bb4, m4, v4, act4);

    // ---- maxpool (NHWC in) -> f32 sequence (flat order == ref reshape) ----
    pool_kernel<<<4096, 256, 0, stream>>>(act4, seq);

    // ---- CfC recurrence, wave-per-sample, LDS-resident masked weights ----
    cfc_rnn_kernel<<<64, 256, 0, stream>>>(
        seq,
        R[0], R[1], R[2], R[3], R[4], R[5], R[6], R[7],
        R[8], R[9], R[10], R[11], R[12], R[13], R[14], R[15],
        R[16], R[17], R[18], R[19], R[20], R[21], R[22], R[23],
        M0, M1, M2, (float*)d_out);
}
